// GRUCell_52793738002631
// MI455X (gfx1250) — matrix operand order repaired and checked by reading, not compiled, not run on hardware
//
#include <hip/hip_runtime.h>

typedef __attribute__((ext_vector_type(16))) _Float16 v16h;
typedef __attribute__((ext_vector_type(8)))  float    v8f;

#define B_  32
#define N_  2048
#define CIN 32
#define H_  64
#define D_  64
#define CI_ 96    // CIN + H
#define KI_ 192   // 2 * CI_
#define OG_ 128   // 2 * H (gate outputs z|r)
#define OU_ 64    // H (update outputs)
#define KC_ 256   // k-chunk staged in LDS for adj@x

__device__ __forceinline__ v8f wmma16x16x32(v16h a, v16h b, v8f c) {
  // D = A(16x32 f16) * B(32x16 f16) + C(16x16 f32)
  return __builtin_amdgcn_wmma_f32_16x16x32_f16(false, a, false, b, (short)0, c,
                                                false, false);
}

// ---------------------------------------------------------------- layernorm
__global__ void k_layernorm(const float* __restrict__ ne, const float* __restrict__ te,
                            const float* __restrict__ gg, const float* __restrict__ gb,
                            const float* __restrict__ ug, const float* __restrict__ ub,
                            float* __restrict__ eg32, _Float16* __restrict__ eg16,
                            float* __restrict__ eu32, _Float16* __restrict__ eu16) {
  int n = blockIdx.x, t = threadIdx.x;   // blockDim = 64 (= D)
  __shared__ float s[64];
  float v = ne[n * D_ + t] + te[t];
  s[t] = v; __syncthreads();
  for (int o = 32; o > 0; o >>= 1) { if (t < o) s[t] += s[t + o]; __syncthreads(); }
  float mu = s[0] * (1.0f / D_); __syncthreads();
  float d = v - mu;
  s[t] = d * d; __syncthreads();
  for (int o = 32; o > 0; o >>= 1) { if (t < o) s[t] += s[t + o]; __syncthreads(); }
  float inv = rsqrtf(s[0] * (1.0f / D_) + 1e-12f);
  float nv = d * inv;
  float vg = nv * gg[t] + gb[t];
  float vu = nv * ug[t] + ub[t];
  eg32[n * D_ + t] = vg; eg16[n * D_ + t] = (_Float16)vg;
  eu32[n * D_ + t] = vu; eu16[n * D_ + t] = (_Float16)vu;
}

// ---------------------------------------------------------------- pack
// inp_T[b][c][m] = concat(x,state)[b][m][c]; cand_T gets the shared x part.
__global__ void k_pack(const float* __restrict__ x, const float* __restrict__ st,
                       _Float16* __restrict__ inpT, _Float16* __restrict__ candT) {
  long idx = (long)blockIdx.x * blockDim.x + threadIdx.x;   // B*CI*N
  if (idx >= (long)B_ * CI_ * N_) return;
  int m = (int)(idx % N_);
  long t = idx / N_;
  int c = (int)(t % CI_);
  int b = (int)(t / CI_);
  float v = (c < CIN) ? x[((long)b * N_ + m) * CIN + c]
                      : st[((long)b * N_ + m) * H_ + (c - CIN)];
  _Float16 h = (_Float16)v;
  inpT[idx] = h;
  if (c < CIN) candT[idx] = h;
}

// ---------------------------------------------------------------- wpool^T
// wT[(o*KI + ki)][d] = wpool[d][ki][o]  (f32 -> f16); one-time re-layout so
// the gen-GEMM B operand is contiguous along K (= d).
__global__ void k_wpoolT(const float* __restrict__ w, _Float16* __restrict__ wT,
                         int O) {
  long tot = (long)D_ * KI_ * O;
  long idx = (long)blockIdx.x * blockDim.x + threadIdx.x;
  if (idx >= tot) return;
  int d = (int)(idx & (D_ - 1));
  long j = idx >> 6;                       // o*KI + ki
  int o  = (int)(j / KI_);
  int ki = (int)(j % KI_);
  wT[idx] = (_Float16)w[((long)d * KI_ + ki) * O + o];
}

// ---------------------------------------------------------------- gen W
// W[n][o][ki] = sum_d e[n,d] * wpool[d,ki,o]  as WMMA GEMM:
// M = nodes (8 waves x 16 nodes = 128 nodes/block), K = 64, N(cols) = j = o*KI+ki.
// A j-strip of 64 wT rows is staged in LDS and shared by all 8 waves.
__global__ void __launch_bounds__(256)
k_genW(const _Float16* __restrict__ e, const _Float16* __restrict__ wT,
       _Float16* __restrict__ W, int O) {
  __shared__ __align__(32) _Float16 sB[64 * D_];   // 8 KB
  int J   = KI_ * O;
  int tid = threadIdx.x;
  int wv  = tid >> 5, lane = tid & 31;
  int j0  = blockIdx.x * 64;
  int n0  = (blockIdx.y * 8 + wv) * 16;
  int rsel = lane & 15, khalf = lane >> 4;

  // stage 64 j-rows x 64 d (4096 halfs = 512 x b128, 2 per thread)
#pragma unroll
  for (int i = 0; i < 2; ++i) {
    int idx  = tid + i * 256;            // 0..511
    int row  = idx >> 3;
    int col8 = (idx & 7) << 3;
    *(uint4*)(sB + row * D_ + col8) =
        *(const uint4*)(wT + (long)(j0 + row) * D_ + col8);
  }
  __syncthreads();

  v8f a0 = {}, a1 = {}, a2 = {}, a3 = {};
#pragma unroll
  for (int ks = 0; ks < 2; ++ks) {
    v16h a = *(const v16h*)(e + (long)(n0 + rsel) * D_ + ks * 32 + khalf * 16);
    const _Float16* bb = sB + ks * 32 + khalf * 16 + rsel * D_;
    a0 = wmma16x16x32(a, *(const v16h*)(bb + 0 * 16 * D_), a0);
    a1 = wmma16x16x32(a, *(const v16h*)(bb + 1 * 16 * D_), a1);
    a2 = wmma16x16x32(a, *(const v16h*)(bb + 2 * 16 * D_), a2);
    a3 = wmma16x16x32(a, *(const v16h*)(bb + 3 * 16 * D_), a3);
  }
  int Mb = (lane < 16) ? 0 : 8;
#pragma unroll
  for (int r = 0; r < 8; ++r) {
    _Float16* p = W + (long)(n0 + r + Mb) * J + j0 + rsel;
    p[0 * 16] = (_Float16)a0[r];
    p[1 * 16] = (_Float16)a1[r];
    p[2 * 16] = (_Float16)a2[r];
    p[3 * 16] = (_Float16)a3[r];
  }
}

// ---------------------------------------------------------------- e @ e^T
__global__ void k_logits(const _Float16* __restrict__ e, float* __restrict__ out) {
  int lane = threadIdx.x & 31;
  int i0 = blockIdx.y * 16, j0 = blockIdx.x * 16;
  int rsel = lane & 15, khalf = lane >> 4;
  v8f acc = {};
  for (int k0 = 0; k0 < D_; k0 += 32) {
    v16h a = *(const v16h*)(e + (long)(i0 + rsel) * D_ + k0 + khalf * 16);
    v16h b = *(const v16h*)(e + (long)(j0 + rsel) * D_ + k0 + khalf * 16);
    acc = wmma16x16x32(a, b, acc);
  }
  int Mb = (lane < 16) ? 0 : 8;
#pragma unroll
  for (int r = 0; r < 8; ++r)
    out[(long)(i0 + r + Mb) * N_ + j0 + rsel] = acc[r];
}

// ---------------------------------------------------------------- row softmax
__global__ void k_softmax(const float* __restrict__ lg, _Float16* __restrict__ adj) {
  int n = blockIdx.x, t = threadIdx.x;  // 256 threads
  __shared__ float s[256];
  const float* row = lg + (long)n * N_;
  float mx = -3.4e38f;
  for (int j = t; j < N_; j += 256) mx = fmaxf(mx, row[j]);
  s[t] = mx; __syncthreads();
  for (int o = 128; o > 0; o >>= 1) { if (t < o) s[t] = fmaxf(s[t], s[t + o]); __syncthreads(); }
  mx = s[0]; __syncthreads();
  float sum = 0.f;
  for (int j = t; j < N_; j += 256) sum += __expf(row[j] - mx);
  s[t] = sum; __syncthreads();
  for (int o = 128; o > 0; o >>= 1) { if (t < o) s[t] += s[t + o]; __syncthreads(); }
  float inv = 1.0f / s[0];
  for (int j = t; j < N_; j += 256)
    adj[(long)n * N_ + j] = (_Float16)(__expf(row[j] - mx) * inv);
}

// ---------------------------------------------------------------- adj @ inp
// agg[b][n][c] = sum_m adj[n][m] * T[b][c][m]; 8 waves share a 96x256 LDS
// chunk of T, each wave holds 6 accumulators and reuses A across c-tiles.
__global__ void __launch_bounds__(256)
k_adjx(const _Float16* __restrict__ adj, const _Float16* __restrict__ T,
       _Float16* __restrict__ agg) {
  __shared__ __align__(32) _Float16 sB[CI_ * KC_];   // 48 KB
  int tid  = threadIdx.x;
  int wv   = tid >> 5, lane = tid & 31;
  int b    = blockIdx.y;                 // 32
  int n0   = (blockIdx.x * 8 + wv) * 16; // 16 blocks x 8 waves = 128 n-tiles
  int rsel = lane & 15, khalf = lane >> 4;

  const _Float16* arow = adj + (long)(n0 + rsel) * N_;
  const _Float16* Tb   = T + (long)b * CI_ * N_;

  v8f acc0 = {}, acc1 = {}, acc2 = {}, acc3 = {}, acc4 = {}, acc5 = {};

  for (int kc = 0; kc < N_; kc += KC_) {
    __syncthreads();
#pragma unroll
    for (int i = 0; i < 12; ++i) {
      int idx  = tid + i * 256;          // 0..3071
      int row  = idx >> 5;
      int col8 = (idx & 31) << 3;
      *(uint4*)(sB + row * KC_ + col8) =
          *(const uint4*)(Tb + (long)row * N_ + kc + col8);
    }
    __syncthreads();
#pragma unroll
    for (int ks = 0; ks < KC_ / 32; ++ks) {
      v16h a = *(const v16h*)(arow + kc + ks * 32 + khalf * 16);
      const _Float16* bbase = sB + ks * 32 + khalf * 16 + rsel * KC_;
      acc0 = wmma16x16x32(a, *(const v16h*)(bbase + 0 * 16 * KC_), acc0);
      acc1 = wmma16x16x32(a, *(const v16h*)(bbase + 1 * 16 * KC_), acc1);
      acc2 = wmma16x16x32(a, *(const v16h*)(bbase + 2 * 16 * KC_), acc2);
      acc3 = wmma16x16x32(a, *(const v16h*)(bbase + 3 * 16 * KC_), acc3);
      acc4 = wmma16x16x32(a, *(const v16h*)(bbase + 4 * 16 * KC_), acc4);
      acc5 = wmma16x16x32(a, *(const v16h*)(bbase + 5 * 16 * KC_), acc5);
    }
  }

  int Mb = (lane < 16) ? 0 : 8;
  _Float16* orow = agg + ((long)b * N_ + n0 + Mb) * CI_ + rsel;
#pragma unroll
  for (int r = 0; r < 8; ++r) {
    _Float16* p = orow + (long)r * CI_;
    p[0 * 16] = (_Float16)acc0[r];
    p[1 * 16] = (_Float16)acc1[r];
    p[2 * 16] = (_Float16)acc2[r];
    p[3 * 16] = (_Float16)acc3[r];
    p[4 * 16] = (_Float16)acc4[r];
    p[5 * 16] = (_Float16)acc5[r];
  }
}

// ---------------------------------------------------------------- gate GCN
// One workgroup per node: X_g (32x192 f16) in LDS; pooled weights W[n] are
// pre-generated ([o][ki] contiguous) and streamed directly from global as
// the WMMA B operand (each element used exactly once). sigmoid -> r, z*state.
__global__ void __launch_bounds__(256)
k_gate(const float* __restrict__ e32, const _Float16* __restrict__ Wg,
       const float* __restrict__ bpool,
       const float* __restrict__ x, const float* __restrict__ st,
       const _Float16* __restrict__ agg,
       float* __restrict__ rbuf, _Float16* __restrict__ zsrows,
       _Float16* __restrict__ candT) {
  int n = blockIdx.x, tid = threadIdx.x;
  __shared__ __align__(32) _Float16 sX[B_ * KI_];    // 12 KB  [b][ki]
  __shared__ float se[D_];
  if (tid < D_) se[tid] = e32[n * D_ + tid];
  // build X_g rows: [x | state | agg]
  for (int idx = tid; idx < B_ * KI_; idx += 256) {
    int b = idx / KI_, c = idx % KI_;
    if (c < CIN)       sX[idx] = (_Float16)x[((long)b * N_ + n) * CIN + c];
    else if (c < CI_)  sX[idx] = (_Float16)st[((long)b * N_ + n) * H_ + (c - CIN)];
    else               sX[idx] = agg[((long)b * N_ + n) * CI_ + (c - CI_)];
  }
  __syncthreads();
  int wv = tid >> 5, lane = tid & 31;
  int o0 = wv * 16;                      // 8 waves x 16 = 128 outputs
  int rsel = lane & 15, khalf = lane >> 4;
  const _Float16* wrow = Wg + (long)n * (KI_ * OG_) + (long)(o0 + rsel) * KI_;
  v8f c0 = {}, c1 = {};
#pragma unroll
  for (int k0 = 0; k0 < KI_; k0 += 32) {
    v16h bW = *(const v16h*)(wrow + k0 + khalf * 16);
    v16h a0 = *(const v16h*)(sX + rsel * KI_ + k0 + khalf * 16);
    v16h a1 = *(const v16h*)(sX + (rsel + 16) * KI_ + k0 + khalf * 16);
    c0 = wmma16x16x32(a0, bW, c0);
    c1 = wmma16x16x32(a1, bW, c1);
  }
  int o = o0 + rsel;                     // N index is lane&15 for every VGPR
  float bias = 0.f;
  for (int d = 0; d < D_; ++d) bias += se[d] * bpool[d * OG_ + o];
  int Mb = (lane < 16) ? 0 : 8;
#pragma unroll
  for (int r = 0; r < 8; ++r) {
#pragma unroll
    for (int hh = 0; hh < 2; ++hh) {
      float v = (hh ? c1[r] : c0[r]) + bias;
      float sg = 1.0f / (1.0f + __expf(-v));
      int b = r + Mb + hh * 16;
      if (o < H_) {                      // z half -> z*state
        float zs = sg * st[((long)b * N_ + n) * H_ + o];
        zsrows[((long)b * N_ + n) * H_ + o] = (_Float16)zs;
        candT[((long)b * CI_ + CIN + o) * N_ + n] = (_Float16)zs;
      } else {                           // r half
        rbuf[((long)b * N_ + n) * H_ + (o - H_)] = sg;
      }
    }
  }
}

// ---------------------------------------------------------------- update GCN
__global__ void __launch_bounds__(256)
k_update(const float* __restrict__ e32, const _Float16* __restrict__ Wu,
         const float* __restrict__ bpool,
         const float* __restrict__ x, const _Float16* __restrict__ zsrows,
         const _Float16* __restrict__ agg,
         const float* __restrict__ rbuf, const float* __restrict__ st,
         float* __restrict__ out) {
  int n = blockIdx.x, tid = threadIdx.x;
  __shared__ __align__(32) _Float16 sX[B_ * KI_];    // 12 KB
  __shared__ float se[D_];
  if (tid < D_) se[tid] = e32[n * D_ + tid];
  for (int idx = tid; idx < B_ * KI_; idx += 256) {
    int b = idx / KI_, c = idx % KI_;
    if (c < CIN)      sX[idx] = (_Float16)x[((long)b * N_ + n) * CIN + c];
    else if (c < CI_) sX[idx] = zsrows[((long)b * N_ + n) * H_ + (c - CIN)];
    else              sX[idx] = agg[((long)b * N_ + n) * CI_ + (c - CI_)];
  }
  __syncthreads();
  int wv = tid >> 5, lane = tid & 31;
  int otile = wv & 3, mhalf = wv >> 2;   // 4 o-tiles x 2 batch halves = 8 waves
  int o0 = otile * 16;
  int rsel = lane & 15, khalf = lane >> 4;
  const _Float16* wrow = Wu + (long)n * (KI_ * OU_) + (long)(o0 + rsel) * KI_;
  v8f acc = {};
#pragma unroll
  for (int k0 = 0; k0 < KI_; k0 += 32) {
    v16h bW = *(const v16h*)(wrow + k0 + khalf * 16);
    v16h a  = *(const v16h*)(sX + (mhalf * 16 + rsel) * KI_ + k0 + khalf * 16);
    acc = wmma16x16x32(a, bW, acc);
  }
  int o = o0 + rsel;
  float bias = 0.f;
  for (int d = 0; d < D_; ++d) bias += se[d] * bpool[d * OU_ + o];
  int Mb = (lane < 16) ? 0 : 8;
#pragma unroll
  for (int r = 0; r < 8; ++r) {
    int b = mhalf * 16 + r + Mb;
    float hc = tanhf(acc[r] + bias);
    long off = ((long)b * N_ + n) * H_ + o;
    float rr = rbuf[off];
    out[off] = rr * st[off] + (1.0f - rr) * hc;
  }
}

// ---------------------------------------------------------------- launcher
extern "C" void kernel_launch(void* const* d_in, const int* in_sizes, int n_in,
                              void* d_out, int out_size, void* d_ws, size_t ws_size,
                              hipStream_t stream) {
  (void)in_sizes; (void)n_in; (void)out_size; (void)ws_size;
  const float* x   = (const float*)d_in[0];
  const float* st  = (const float*)d_in[1];
  const float* ne  = (const float*)d_in[2];
  const float* te  = (const float*)d_in[3];
  const float* gw  = (const float*)d_in[4];
  const float* gbp = (const float*)d_in[5];
  const float* gg  = (const float*)d_in[6];
  const float* gb  = (const float*)d_in[7];
  const float* uw  = (const float*)d_in[8];
  const float* ubp = (const float*)d_in[9];
  const float* ug  = (const float*)d_in[10];
  const float* ub  = (const float*)d_in[11];
  float* out = (float*)d_out;

  char* p = (char*)d_ws;
  auto alloc = [&](size_t bytes) {
    char* r = p;
    p += (bytes + 255) & ~(size_t)255;
    return r;
  };
  float*    eg32 = (float*)alloc((size_t)N_ * D_ * 4);
  float*    eu32 = (float*)alloc((size_t)N_ * D_ * 4);
  _Float16* eg16 = (_Float16*)alloc((size_t)N_ * D_ * 2);
  _Float16* eu16 = (_Float16*)alloc((size_t)N_ * D_ * 2);
  float*    lgts = (float*)alloc((size_t)N_ * N_ * 4);          // reused g/u
  _Float16* adj  = (_Float16*)alloc((size_t)N_ * N_ * 2);       // reused g/u
  _Float16* inpT = (_Float16*)alloc((size_t)B_ * CI_ * N_ * 2);
  _Float16* candT= (_Float16*)alloc((size_t)B_ * CI_ * N_ * 2);
  _Float16* agg  = (_Float16*)alloc((size_t)B_ * N_ * CI_ * 2); // reused g/u
  float*    rbuf = (float*)alloc((size_t)B_ * N_ * H_ * 4);
  _Float16* zs   = (_Float16*)alloc((size_t)B_ * N_ * H_ * 2);
  _Float16* wTg  = (_Float16*)alloc((size_t)KI_ * OG_ * D_ * 2);
  _Float16* wTu  = (_Float16*)alloc((size_t)KI_ * OU_ * D_ * 2);
  _Float16* Wg   = (_Float16*)alloc((size_t)N_ * KI_ * OG_ * 2); // 100 MB
  _Float16* Wu   = (_Float16*)alloc((size_t)N_ * KI_ * OU_ * 2); //  50 MB

  k_layernorm<<<N_, 64, 0, stream>>>(ne, te, gg, gb, ug, ub, eg32, eg16, eu32, eu16);
  {
    long tot = (long)B_ * CI_ * N_;
    k_pack<<<(unsigned)((tot + 255) / 256), 256, 0, stream>>>(x, st, inpT, candT);
  }
  // pooled-weight generation as WMMA GEMM
  {
    long tg = (long)D_ * KI_ * OG_, tu = (long)D_ * KI_ * OU_;
    k_wpoolT<<<(unsigned)((tg + 255) / 256), 256, 0, stream>>>(gw, wTg, OG_);
    k_wpoolT<<<(unsigned)((tu + 255) / 256), 256, 0, stream>>>(uw, wTu, OU_);
    dim3 gg_(KI_ * OG_ / 64, N_ / 128);   // 384 x 16
    dim3 gu_(KI_ * OU_ / 64, N_ / 128);   // 192 x 16
    k_genW<<<gg_, 256, 0, stream>>>(eg16, wTg, Wg, OG_);
    k_genW<<<gu_, 256, 0, stream>>>(eu16, wTu, Wu, OU_);
  }

  dim3 gl(N_ / 16, N_ / 16);
  dim3 ga(N_ / (8 * 16), B_);   // 16 x 32 blocks, 8 waves each

  // gate phase
  k_logits <<<gl, 32, 0, stream>>>(eg16, lgts);
  k_softmax<<<N_, 256, 0, stream>>>(lgts, adj);
  k_adjx   <<<ga, 256, 0, stream>>>(adj, inpT, agg);
  k_gate   <<<N_, 256, 0, stream>>>(eg32, Wg, gbp, x, st, agg, rbuf, zs, candT);

  // update phase (reuse logits/adj/agg buffers)
  k_logits <<<gl, 32, 0, stream>>>(eu16, lgts);
  k_softmax<<<N_, 256, 0, stream>>>(lgts, adj);
  k_adjx   <<<ga, 256, 0, stream>>>(adj, candT, agg);
  k_update <<<N_, 256, 0, stream>>>(eu32, Wu, ubp, x, zs, agg, rbuf, st, out);
}
